// CrossAttention_71983651881635
// MI455X (gfx1250) — matrix operand order repaired and checked by reading, not compiled, not run on hardware
//
#include <hip/hip_runtime.h>
#include <hip/hip_bf16.h>

typedef __attribute__((ext_vector_type(16))) _Float16 v16h;
typedef __attribute__((ext_vector_type(8)))  _Float16 v8h;
typedef __attribute__((ext_vector_type(8)))  float    v8f;

#define DIM      768
#define HEADS    8
#define HEAD_DIM 96
#define SEQ      1024
#define BATCH    8
#define ROWS     (BATCH * SEQ)          // 8192
#define SCALE    0.10206207261596575f   // 96^-0.5
#define LDJ      40                     // Kt row stride (halves); 80 B, 16B aligned
#define LDB      136                    // Bs row stride (halves); 272 B, 16B aligned, bank-spread

// LDS byte offset of a __shared__ pointer (flat LDS-aperture addr low 32 bits)
__device__ __forceinline__ unsigned lds_off(const void* p) {
    return (unsigned)(unsigned long long)p;
}
// Async global->LDS 16B copy (tracked by ASYNCcnt), GV addressing mode.
__device__ __forceinline__ void async_copy16(unsigned lds_byte, const void* gptr) {
    asm volatile("global_load_async_to_lds_b128 %0, %1, off"
                 :: "v"(lds_byte), "v"((unsigned long long)gptr) : "memory");
}
__device__ __forceinline__ void async_wait0() {
    asm volatile("s_wait_asynccnt 0x0" ::: "memory");
}

// ---------------------------------------------------------------------------
// LayerNorm (row of 768) -> f16. One 256-thread block per row (8 waves).
// ---------------------------------------------------------------------------
__global__ void layernorm_f16(const float* __restrict__ X,
                              const float* __restrict__ gam,
                              const float* __restrict__ bet,
                              _Float16* __restrict__ Y) {
    const int r = blockIdx.x, t = threadIdx.x;
    const float* xr = X + (size_t)r * DIM;
    float x0 = xr[t], x1 = xr[t + 256], x2 = xr[t + 512];

    __shared__ float red[8];
    float s = x0 + x1 + x2;
#pragma unroll
    for (int m = 16; m >= 1; m >>= 1) s += __shfl_xor(s, m, 32);
    if ((t & 31) == 0) red[t >> 5] = s;
    __syncthreads();
    float tot = 0.f;
#pragma unroll
    for (int i = 0; i < 8; ++i) tot += red[i];
    const float mu = tot * (1.0f / DIM);

    float d0 = x0 - mu, d1 = x1 - mu, d2 = x2 - mu;
    float v = d0 * d0 + d1 * d1 + d2 * d2;
#pragma unroll
    for (int m = 16; m >= 1; m >>= 1) v += __shfl_xor(v, m, 32);
    __syncthreads();
    if ((t & 31) == 0) red[t >> 5] = v;
    __syncthreads();
    float vt = 0.f;
#pragma unroll
    for (int i = 0; i < 8; ++i) vt += red[i];
    const float rstd = rsqrtf(vt * (1.0f / DIM) + 1e-5f);

    _Float16* yr = Y + (size_t)r * DIM;
    yr[t]       = (_Float16)(d0 * rstd * gam[t]       + bet[t]);
    yr[t + 256] = (_Float16)(d1 * rstd * gam[t + 256] + bet[t + 256]);
    yr[t + 512] = (_Float16)(d2 * rstd * gam[t + 512] + bet[t + 512]);
}

// ---------------------------------------------------------------------------
__global__ void f32_to_f16(const float* __restrict__ s, _Float16* __restrict__ d, int n) {
    int i = blockIdx.x * blockDim.x + threadIdx.x;
    const int stride = gridDim.x * blockDim.x;
    for (; i < n; i += stride) d[i] = (_Float16)s[i];
}

__global__ void zero_f32(float* __restrict__ p, int n) {
    int i = blockIdx.x * blockDim.x + threadIdx.x;
    if (i < n) p[i] = 0.f;
}

// ---------------------------------------------------------------------------
// WMMA GEMM: C[M,N] = A[M,K]*B[K,N] (+bias)(+residual), f16 in, f16/f32 out.
// Block tile 64x128 (8 waves, each 16x64). B tile (32 x 128) double-buffered
// in LDS via async global->LDS b128 copies; next tile staged during WMMAs.
// A frag: lane l holds row l%16, k-chunks [8*(l/16),+8) and +16 (ISA 7.12.2).
// B frag: lane l holds column n0+l%16, same k order as A -> strided down LDS rows.
// ---------------------------------------------------------------------------
template <bool OUT_F16>
__global__ __launch_bounds__(256, 2)
void gemm_wmma(const _Float16* __restrict__ A,
               const _Float16* __restrict__ Bw,
               const float* __restrict__ bias,
               void* __restrict__ Cout,
               const float* __restrict__ residual,
               int M, int N, int K) {
    const int tid  = threadIdx.x;
    const int w    = tid >> 5, lane = tid & 31;
    const int l16  = lane & 15, hi = lane >> 4;
    const int mi   = w >> 1, ni = w & 1;
    const int row0 = blockIdx.x * 64 + mi * 16;
    const int colB = blockIdx.y * 128;           // block's N origin

    __shared__ _Float16 Bs[2][32 * LDB];         // double-buffered [k][n]

    const int kk = tid >> 4, np = (tid & 15) * 8;      // this thread's 2 chunks
    const int kk2 = (tid + 256) >> 4, np2 = ((tid + 256) & 15) * 8;
    auto stage = [&](int k0s, int bufs) {
        const _Float16* g0 = Bw + (size_t)(k0s + kk) * N + colB + np;
        const _Float16* g1 = Bw + (size_t)(k0s + kk2) * N + colB + np2;
        async_copy16(lds_off(&Bs[bufs][kk * LDB + np]),  g0);
        async_copy16(lds_off(&Bs[bufs][kk2 * LDB + np2]), g1);
    };

    v8f acc[4] = {};
    const _Float16* arow = A + (size_t)(row0 + l16) * K;

    stage(0, 0);
    async_wait0();
    __syncthreads();

    for (int k0 = 0; k0 < K; k0 += 32) {
        const int buf = (k0 >> 5) & 1;
        if (k0 + 32 < K) stage(k0 + 32, buf ^ 1);   // overlap with WMMAs

        v16h a;
        const int kb = k0 + hi * 8;
        *((v8h*)&a)     = *(const v8h*)(arow + kb);
        *((v8h*)&a + 1) = *(const v8h*)(arow + kb + 16);

#pragma unroll
        for (int t = 0; t < 4; ++t) {
            v16h bfr;
            const _Float16* bp = &Bs[buf][hi * 8 * LDB + ni * 64 + t * 16 + l16];
#pragma unroll
            for (int s = 0; s < 16; ++s) bfr[s] = bp[(s < 8 ? s : s + 8) * LDB];
            acc[t] = __builtin_amdgcn_wmma_f32_16x16x32_f16(
                false, a, false, bfr, (short)0, acc[t], false, false);
        }
        async_wait0();      // next buffer staged
        __syncthreads();    // all waves done reading buf; buf^1 visible
    }

#pragma unroll
    for (int t = 0; t < 4; ++t) {
#pragma unroll
        for (int r = 0; r < 8; ++r) {
            const int row = row0 + r + hi * 8;
            const int col = colB + ni * 64 + t * 16 + l16;
            float v = acc[t][r] + bias[col];
            if (residual) v += residual[(size_t)row * N + col];
            if (OUT_F16) ((_Float16*)Cout)[(size_t)row * N + col] = (_Float16)v;
            else         ((float*)Cout)[(size_t)row * N + col] = v;
        }
    }
}

// ---------------------------------------------------------------------------
// Flash attention per (b, h, 128-row q tile). 8 waves x 16 q rows.
// Double-buffered K (transposed [d][j]) and V (row-major, async copies) tiles;
// one barrier per j-step. Probs bounce through a per-wave LDS tile (C-layout
// f32 -> A-layout f16); same-wave LDS ordering makes that barrier-free.
// ---------------------------------------------------------------------------
__global__ __launch_bounds__(256, 2)
void flash_attn(const _Float16* __restrict__ Qh,
                const _Float16* __restrict__ KVh,
                _Float16* __restrict__ Oh,
                float* __restrict__ map) {
    const int b = blockIdx.z, h = blockIdx.y, qt = blockIdx.x;
    const int tid = threadIdx.x;
    const int w = tid >> 5, lane = tid & 31;
    const int l16 = lane & 15, hi = lane >> 4;

    __shared__ _Float16 Kt[2][HEAD_DIM * LDJ];   // transposed K tile [d][j]
    __shared__ _Float16 Vs[2][32 * HEAD_DIM];    // V tile row-major [j][d]
    __shared__ _Float16 Ps[8][16 * 32];          // per-wave prob tile

    const int qrow0 = qt * 128 + w * 16;         // local row inside batch b
    const _Float16* kvb = KVh + (size_t)b * SEQ * (2 * DIM) + h * HEAD_DIM;
    _Float16* pw = Ps[w];

    // stage one 32-row K/V tile (V async, K transposed through registers)
    auto stage = [&](int j0s, int bufs) {
        for (int c = tid; c < 384; c += 256) {
            const int row = c / 12, dp = (c % 12) * 8;
            const _Float16* kp = kvb + (size_t)(j0s + row) * (2 * DIM) + dp;
            async_copy16(lds_off(&Vs[bufs][row * HEAD_DIM + dp]), kp + DIM);
            v8h k8 = *(const v8h*)kp;
#pragma unroll
            for (int i = 0; i < 8; ++i) Kt[bufs][(dp + i) * LDJ + row] = k8[i];
        }
    };

    // Q fragments (persist in registers): 16 rows x 96 = 3 chunks of K=32
    v16h aq[3];
    {
        const _Float16* qb = Qh + (size_t)(b * SEQ + qrow0 + l16) * DIM + h * HEAD_DIM;
#pragma unroll
        for (int kc = 0; kc < 3; ++kc) {
            const int kb = kc * 32 + hi * 8;
            *((v8h*)&aq[kc])     = *(const v8h*)(qb + kb);
            *((v8h*)&aq[kc] + 1) = *(const v8h*)(qb + kb + 16);
        }
    }

    v8f o[6] = {};
    float mrow[8], lrow[8];
#pragma unroll
    for (int r = 0; r < 8; ++r) { mrow[r] = -1e30f; lrow[r] = 0.f; }

    stage(0, 0);
    async_wait0();
    __syncthreads();

    for (int j0 = 0; j0 < SEQ; j0 += 32) {
        const int buf = (j0 >> 5) & 1;
        if (j0 + 32 < SEQ) stage(j0 + 32, buf ^ 1);   // overlap with compute

        // ---- S = Q K^T (16x32) : 6 WMMA
        v8f s[2];
#pragma unroll
        for (int jt = 0; jt < 2; ++jt) {
            v8f a2 = {};
#pragma unroll
            for (int kc = 0; kc < 3; ++kc) {
                v16h bfr;
                const _Float16* kb = &Kt[buf][(kc * 32 + hi * 8) * LDJ + jt * 16 + l16];
#pragma unroll
                for (int s = 0; s < 16; ++s) bfr[s] = kb[(s < 8 ? s : s + 8) * LDJ];
                a2 = __builtin_amdgcn_wmma_f32_16x16x32_f16(
                    false, aq[kc], false, bfr, (short)0, a2, false, false);
            }
            s[jt] = a2;
        }

        // ---- online softmax; probs written straight to per-wave LDS tile
        float corr[8];
#pragma unroll
        for (int r = 0; r < 8; ++r) {
            const float a0 = s[0][r] * SCALE, a1 = s[1][r] * SCALE;
            float mx = fmaxf(a0, a1);
            mx = fmaxf(mx, __shfl_xor(mx, 1, 32));
            mx = fmaxf(mx, __shfl_xor(mx, 2, 32));
            mx = fmaxf(mx, __shfl_xor(mx, 4, 32));
            mx = fmaxf(mx, __shfl_xor(mx, 8, 32));
            const float mn = fmaxf(mrow[r], mx);
            corr[r] = __expf(mrow[r] - mn);
            const float p0 = __expf(a0 - mn);
            const float p1 = __expf(a1 - mn);
            pw[(r + hi * 8) * 32 + l16]      = (_Float16)p0;
            pw[(r + hi * 8) * 32 + 16 + l16] = (_Float16)p1;
            float rs = p0 + p1;
            rs += __shfl_xor(rs, 1, 32);
            rs += __shfl_xor(rs, 2, 32);
            rs += __shfl_xor(rs, 4, 32);
            rs += __shfl_xor(rs, 8, 32);
            lrow[r] = lrow[r] * corr[r] + rs;
            mrow[r] = mn;
        }
#pragma unroll
        for (int nt = 0; nt < 6; ++nt)
#pragma unroll
            for (int r = 0; r < 8; ++r) o[nt][r] *= corr[r];

        // ---- O += P V : 6 WMMA (Ps: same-wave LDS ordering, no barrier)
        v16h aw;
        {
            const int kb = hi * 8;
            *((v8h*)&aw)     = *(const v8h*)(pw + l16 * 32 + kb);
            *((v8h*)&aw + 1) = *(const v8h*)(pw + l16 * 32 + kb + 16);
        }
#pragma unroll
        for (int nt = 0; nt < 6; ++nt) {
            v16h bv;
            const _Float16* vb = &Vs[buf][hi * 8 * HEAD_DIM + nt * 16 + l16];
#pragma unroll
            for (int s = 0; s < 16; ++s) bv[s] = vb[(s < 8 ? s : s + 8) * HEAD_DIM];
            o[nt] = __builtin_amdgcn_wmma_f32_16x16x32_f16(
                false, aw, false, bv, (short)0, o[nt], false, false);
        }

        async_wait0();      // next-tile V copies landed
        __syncthreads();    // all waves done with buf; buf^1 fully staged
    }

    // ---- finalize: O /= l, write f16; accumulate attn_map
    float inv[8];
#pragma unroll
    for (int r = 0; r < 8; ++r) inv[r] = 1.0f / lrow[r];
    _Float16* ob = Oh + (size_t)(b * SEQ + qrow0) * DIM + h * HEAD_DIM;
#pragma unroll
    for (int nt = 0; nt < 6; ++nt)
#pragma unroll
        for (int r = 0; r < 8; ++r) {
            const int row = r + hi * 8;
            ob[(size_t)row * DIM + nt * 16 + l16] = (_Float16)(o[nt][r] * inv[r]);
        }
    if (l16 == 0) {
        // sum_m softmax == 1 per row; mean over 8 heads -> each head adds 1/8
#pragma unroll
        for (int r = 0; r < 8; ++r)
            atomicAdd(map + b * SEQ + qrow0 + r + hi * 8, 0.125f);
    }
}

// ---------------------------------------------------------------------------
extern "C" void kernel_launch(void* const* d_in, const int* in_sizes, int n_in,
                              void* d_out, int out_size, void* d_ws, size_t ws_size,
                              hipStream_t stream) {
    const float* x   = (const float*)d_in[0];
    const float* ctx = (const float*)d_in[1];
    const float* Wq  = (const float*)d_in[2];
    const float* bq  = (const float*)d_in[3];
    const float* Wkv = (const float*)d_in[4];
    const float* bkv = (const float*)d_in[5];
    const float* Wp  = (const float*)d_in[6];
    const float* bp  = (const float*)d_in[7];
    const float* g1  = (const float*)d_in[8];
    const float* b1  = (const float*)d_in[9];
    const float* g2  = (const float*)d_in[10];
    const float* b2  = (const float*)d_in[11];

    float* out = (float*)d_out;
    float* map = out + (size_t)ROWS * DIM;   // attn_map: 8*32*32 floats

    char* p = (char*)d_ws;
    _Float16* xn   = (_Float16*)p; p += (size_t)ROWS * DIM * 2;
    _Float16* cn   = (_Float16*)p; p += (size_t)ROWS * DIM * 2;
    _Float16* WqH  = (_Float16*)p; p += (size_t)DIM * DIM * 2;
    _Float16* WkvH = (_Float16*)p; p += (size_t)DIM * 2 * DIM * 2;
    _Float16* WpH  = (_Float16*)p; p += (size_t)DIM * DIM * 2;
    _Float16* Qb   = (_Float16*)p; p += (size_t)ROWS * DIM * 2;
    _Float16* KVb  = (_Float16*)p; p += (size_t)ROWS * 2 * DIM * 2;
    _Float16* Ob   = (_Float16*)p; p += (size_t)ROWS * DIM * 2;

    // 1) LayerNorms -> f16
    layernorm_f16<<<ROWS, 256, 0, stream>>>(x,   g1, b1, xn);
    layernorm_f16<<<ROWS, 256, 0, stream>>>(ctx, g2, b2, cn);

    // 2) Weights -> f16
    f32_to_f16<<<512, 256, 0, stream>>>(Wq,  WqH,  DIM * DIM);
    f32_to_f16<<<512, 256, 0, stream>>>(Wkv, WkvH, DIM * 2 * DIM);
    f32_to_f16<<<512, 256, 0, stream>>>(Wp,  WpH,  DIM * DIM);

    // 3) Projections (WMMA GEMMs): Q = xn*Wq+bq ; KV = cn*Wkv+bkv
    gemm_wmma<true><<<dim3(ROWS / 64, DIM / 128), 256, 0, stream>>>(
        xn, WqH, bq, (void*)Qb, nullptr, ROWS, DIM, DIM);
    gemm_wmma<true><<<dim3(ROWS / 64, (2 * DIM) / 128), 256, 0, stream>>>(
        cn, WkvH, bkv, (void*)KVb, nullptr, ROWS, 2 * DIM, DIM);

    // 4) Attention (flash, WMMA) + attn_map accumulation
    zero_f32<<<32, 256, 0, stream>>>(map, BATCH * SEQ);
    flash_attn<<<dim3(SEQ / 128, HEADS, BATCH), 256, 0, stream>>>(Qb, KVb, Ob, map);

    // 5) Output projection + bias + residual -> fp32 d_out
    gemm_wmma<false><<<dim3(ROWS / 64, DIM / 128), 256, 0, stream>>>(
        Ob, WpH, bp, (void*)out, x, ROWS, DIM, DIM);
}
